// DualModel_3908420240157
// MI455X (gfx1250) — matrix operand order repaired and checked by reading, not compiled, not run on hardware
//
#include <hip/hip_runtime.h>
#include <hip/hip_bf16.h>

typedef __bf16 bf16_t;
typedef __attribute__((ext_vector_type(16))) __bf16 v16bf;
typedef __attribute__((ext_vector_type(8)))  __bf16 v8bfv;
typedef __attribute__((ext_vector_type(8)))  float  v8f;

#define NN   10000
#define IND  512
#define HIDD 512
#define OUTD 256
#define EE   320000

// ---------------------------------------------------------------------------
// WMMA fragment helpers (bf16, 16x16x32, layouts per CDNA5 ISA)
// ---------------------------------------------------------------------------

// A-matrix 16x32 (MxK): lane L holds row m = L&15.
// elems 0..7  -> K = k0 + (L&16 ? 8 : 0) + i
// elems 8..15 -> K = k0 + (L&16 ? 8 : 0) + 16 + (i-8)
__device__ __forceinline__ v16bf load_fragA(const bf16_t* __restrict__ base, int ld,
                                            int row0, int k0, int lane) {
  int r  = row0 + (lane & 15);
  int kb = k0 + ((lane & 16) ? 8 : 0);
  const bf16_t* p = base + (size_t)r * ld + kb;
  v8bfv lo = *(const v8bfv*)(p);
  v8bfv hi = *(const v8bfv*)(p + 16);
  v16bf f;
#pragma unroll
  for (int i = 0; i < 8; ++i) { f[i] = lo[i]; f[i + 8] = hi[i]; }
  return f;
}

// B-matrix 32x16 (KxN) sourced from Bt (N x K row-major): lane L holds col n = L&15,
// elems 0..7 -> K = k0 + (L&16 ? 8 : 0) + i, elems 8..15 -> K = k0 + (L&16 ? 8 : 0) + 16 + (i-8).
__device__ __forceinline__ v16bf load_fragB(const bf16_t* __restrict__ baseT, int ld,
                                            int col0, int k0, int lane) {
  int r  = col0 + (lane & 15);
  int kb = k0 + ((lane & 16) ? 8 : 0);
  const bf16_t* p = baseT + (size_t)r * ld + kb;
  v8bfv lo = *(const v8bfv*)(p);
  v8bfv hi = *(const v8bfv*)(p + 16);
  v16bf f;
#pragma unroll
  for (int i = 0; i < 8; ++i) { f[i] = lo[i]; f[i + 8] = hi[i]; }
  return f;
}

// ---------------------------------------------------------------------------
// Dense GEMM: C[M x Ncols] = A[M x K] * Bt[Ncols x K]^T (+bias) (+relu)
// block = 256 threads = 8 waves, each wave one 16x16 tile (2 rows x 4 cols)
// ---------------------------------------------------------------------------
__global__ void wmma_gemm(const bf16_t* __restrict__ A, const bf16_t* __restrict__ Bt,
                          float* __restrict__ C, bf16_t* __restrict__ Cbf,
                          const float* __restrict__ bias,
                          int M, int Ncols, int K, int relu) {
  const int lane = threadIdx.x & 31;
  const int wave = threadIdx.x >> 5;
  int tileM = blockIdx.y * 2 + (wave >> 2);
  int tileN = blockIdx.x * 4 + (wave & 3);
  if (tileM * 16 >= M || tileN * 16 >= Ncols) return;   // wave-uniform
  int row0 = tileM * 16, col0 = tileN * 16;

  v8f acc = {};
  for (int k0 = 0; k0 < K; k0 += 32) {
    v16bf a = load_fragA(A, K, row0, k0, lane);
    v16bf b = load_fragB(Bt, K, col0, k0, lane);
    acc = __builtin_amdgcn_wmma_f32_16x16x32_bf16(false, a, false, b,
                                                  (short)0, acc, false, false);
  }
  int n  = col0 + (lane & 15);
  int mb = row0 + ((lane & 16) ? 8 : 0);
  float bv = bias ? bias[n] : 0.0f;
#pragma unroll
  for (int r = 0; r < 8; ++r) {
    float v = acc[r] + bv;
    if (relu) v = v > 0.0f ? v : 0.0f;
    size_t o = (size_t)(mb + r) * Ncols + n;
    if (C)   C[o]   = v;
    if (Cbf) Cbf[o] = (bf16_t)v;
  }
}

// ---------------------------------------------------------------------------
// Fused N x N similarity GEMM with exp()/row-sum epilogue:
//   rowNeg[i] += sum_j exp( (Z[i,:] . Bt[j,:]) / tau )
// Each wave owns one 16-row tile; A fragments (K=256 -> 8 frags) hoisted.
// ---------------------------------------------------------------------------
__global__ void k_neg_rowsum(const bf16_t* __restrict__ Z, const bf16_t* __restrict__ Bt,
                             float* __restrict__ rowNeg, const float* __restrict__ tau,
                             int nRowTiles, int nColTiles, int chunkLen) {
  const int lane = threadIdx.x & 31;
  const int wave = threadIdx.x >> 5;
  int rowTile = blockIdx.y * 8 + wave;
  if (rowTile >= nRowTiles) return;                     // wave-uniform
  int row0 = rowTile * 16;
  float invtau = 1.0f / tau[0];

  v16bf afrag[8];
#pragma unroll
  for (int kk = 0; kk < 8; ++kk)
    afrag[kk] = load_fragA(Z, OUTD, row0, kk * 32, lane);

  float rsum[8];
#pragma unroll
  for (int r = 0; r < 8; ++r) rsum[r] = 0.0f;

  int ct0 = blockIdx.x * chunkLen;
  int ct1 = ct0 + chunkLen; if (ct1 > nColTiles) ct1 = nColTiles;
  for (int ct = ct0; ct < ct1; ++ct) {
    int col0 = ct * 16;
    v8f acc = {};
#pragma unroll
    for (int kk = 0; kk < 8; ++kk) {
      v16bf b = load_fragB(Bt, OUTD, col0, kk * 32, lane);
      acc = __builtin_amdgcn_wmma_f32_16x16x32_bf16(false, afrag[kk], false, b,
                                                    (short)0, acc, false, false);
    }
#pragma unroll
    for (int r = 0; r < 8; ++r) rsum[r] += __expf(acc[r] * invtau);
  }
  // reduce across the 16 lanes of each half-wave (the n dimension)
#pragma unroll
  for (int r = 0; r < 8; ++r) {
    float s = rsum[r];
    s += __shfl_xor(s, 1, 32);
    s += __shfl_xor(s, 2, 32);
    s += __shfl_xor(s, 4, 32);
    s += __shfl_xor(s, 8, 32);
    rsum[r] = s;
  }
  if (lane == 0 || lane == 16) {
    int mb = row0 + ((lane & 16) ? 8 : 0);
#pragma unroll
    for (int r = 0; r < 8; ++r) atomicAdd(&rowNeg[mb + r], rsum[r]);
  }
}

// ---------------------------------------------------------------------------
// Graph / elementwise kernels
// ---------------------------------------------------------------------------
__global__ void k_deg(const int* __restrict__ idx, float* __restrict__ deg, int E) {
  int e = blockIdx.x * blockDim.x + threadIdx.x;
  if (e < E) atomicAdd(&deg[idx[e]], 1.0f);
}

__global__ void k_invsqrt(float* __restrict__ d, int n) {
  int i = blockIdx.x * blockDim.x + threadIdx.x;
  if (i < n) { float v = d[i]; d[i] = v > 0.0f ? rsqrtf(v) : 0.0f; }
}

__global__ void k_f32_to_bf16(const float* __restrict__ x, bf16_t* __restrict__ y, long long n) {
  long long i = (long long)blockIdx.x * blockDim.x + threadIdx.x;
  if (i < n) y[i] = (bf16_t)x[i];
}

// W is K x Mo row-major fp32; Wt is Mo x K row-major bf16 (W^T)
__global__ void k_transpose_bf16(const float* __restrict__ W, bf16_t* __restrict__ Wt,
                                 int K, int Mo) {
  int i = blockIdx.x * blockDim.x + threadIdx.x;
  if (i >= K * Mo) return;
  int k = i / Mo, m = i - k * Mo;
  Wt[(size_t)m * K + k] = (bf16_t)W[i];
}

// AGG[dst] += H[src] * ns[src], 4 floats per thread
__global__ void k_scatter_agg(const float* __restrict__ H, const float* __restrict__ ns,
                              const int* __restrict__ src, const int* __restrict__ dst,
                              float* __restrict__ AGG, long long total, int F, int cshift) {
  long long tid = (long long)blockIdx.x * blockDim.x + threadIdx.x;
  if (tid >= total) return;
  int e = (int)(tid >> cshift);
  int c = (int)(tid & ((1 << cshift) - 1));
  int s = src[e], d = dst[e];
  float w = ns[s];
  float4 h = ((const float4*)(H + (size_t)s * F))[c];
  float* o = AGG + (size_t)d * F + (c << 2);
  atomicAdd(o + 0, h.x * w);
  atomicAdd(o + 1, h.y * w);
  atomicAdd(o + 2, h.z * w);
  atomicAdd(o + 3, h.w * w);
}

__global__ void k_finalize(const float* __restrict__ AGG, const float* __restrict__ nd,
                           const float* __restrict__ bias, float* __restrict__ outF,
                           bf16_t* __restrict__ outBf, long long total,
                           int F, int fshift, int relu) {
  long long idx = (long long)blockIdx.x * blockDim.x + threadIdx.x;
  if (idx >= total) return;
  int f = (int)(idx & (F - 1));
  int i = (int)(idx >> fshift);
  float v = AGG[idx] * nd[i] + bias[f];
  if (relu) v = v > 0.0f ? v : 0.0f;
  if (outF)  outF[idx]  = v;
  if (outBf) outBf[idx] = (bf16_t)v;
}

// Row-wise L2 normalize, D = 256, one 256-thread block per row
__global__ void k_rownorm(const float* __restrict__ X, float* __restrict__ outF,
                          bf16_t* __restrict__ outBf) {
  int row = blockIdx.x;
  size_t o = (size_t)row * OUTD + threadIdx.x;
  float v = X[o];
  float ss = v * v;
  ss += __shfl_xor(ss, 1, 32);  ss += __shfl_xor(ss, 2, 32);
  ss += __shfl_xor(ss, 4, 32);  ss += __shfl_xor(ss, 8, 32);
  ss += __shfl_xor(ss, 16, 32);
  __shared__ float sm[8];
  if ((threadIdx.x & 31) == 0) sm[threadIdx.x >> 5] = ss;
  __syncthreads();
  float tot = 0.0f;
#pragma unroll
  for (int i = 0; i < 8; ++i) tot += sm[i];
  float inv = 1.0f / fmaxf(sqrtf(tot), 1e-12f);
  float r = v * inv;
  if (outF)  outF[o]  = r;
  if (outBf) outBf[o] = (bf16_t)r;
}

// one wave per edge: sim[e] = <un[src], q[dst]> / tau; posnum[dst] += sim
__global__ void k_sim_pos(const float* __restrict__ un, const float* __restrict__ q,
                          const int* __restrict__ src, const int* __restrict__ dst,
                          const float* __restrict__ tau, float* __restrict__ sim,
                          float* __restrict__ posnum, int E) {
  int wid  = (int)(((long long)blockIdx.x * blockDim.x + threadIdx.x) >> 5);
  int lane = threadIdx.x & 31;
  if (wid >= E) return;
  int s = src[wid], d = dst[wid];
  const float4* a = (const float4*)(un + (size_t)s * OUTD);
  const float4* b = (const float4*)(q  + (size_t)d * OUTD);
  float acc = 0.0f;
#pragma unroll
  for (int i = 0; i < 2; ++i) {
    float4 av = a[lane + 32 * i], bv = b[lane + 32 * i];
    acc += av.x * bv.x + av.y * bv.y + av.z * bv.z + av.w * bv.w;
  }
  acc += __shfl_xor(acc, 1, 32);  acc += __shfl_xor(acc, 2, 32);
  acc += __shfl_xor(acc, 4, 32);  acc += __shfl_xor(acc, 8, 32);
  acc += __shfl_xor(acc, 16, 32);
  if (lane == 0) {
    float sv = acc / tau[0];
    sim[wid] = sv;
    atomicAdd(&posnum[d], sv);
  }
}

__global__ void k_neg_edge(const float* __restrict__ rowNeg, const float* __restrict__ sim,
                           const int* __restrict__ dst, float* __restrict__ negnum, int E) {
  int e = blockIdx.x * blockDim.x + threadIdx.x;
  if (e >= E) return;
  int d = dst[e];
  float m = logf(rowNeg[d] + __expf(sim[e]));
  atomicAdd(&negnum[d], m);
}

__global__ void k_loss(const float* __restrict__ pos, const float* __restrict__ neg,
                       const float* __restrict__ cnt, float* __restrict__ loss, int n) {
  float s = 0.0f;
  for (int i = blockIdx.x * blockDim.x + threadIdx.x; i < n; i += gridDim.x * blockDim.x) {
    float d = fmaxf(cnt[i], 1.0f);
    s += (neg[i] - pos[i]) / d;
  }
  s += __shfl_xor(s, 1, 32);  s += __shfl_xor(s, 2, 32);
  s += __shfl_xor(s, 4, 32);  s += __shfl_xor(s, 8, 32);
  s += __shfl_xor(s, 16, 32);
  __shared__ float sm[8];
  if ((threadIdx.x & 31) == 0) sm[threadIdx.x >> 5] = s;
  __syncthreads();
  if (threadIdx.x < 8) {
    float t = sm[threadIdx.x];
    t += __shfl_xor(t, 1, 32);
    t += __shfl_xor(t, 2, 32);
    t += __shfl_xor(t, 4, 32);
    if (threadIdx.x == 0) atomicAdd(loss, t / (float)n);
  }
}

// ---------------------------------------------------------------------------
// Host-side orchestration helpers
// ---------------------------------------------------------------------------
static void launch_gemm(const bf16_t* A, const bf16_t* Bt, float* C, bf16_t* Cbf,
                        const float* bias, int M, int Ncols, int K, int relu,
                        hipStream_t st) {
  dim3 grid((Ncols + 63) / 64, (M + 31) / 32);
  wmma_gemm<<<grid, 256, 0, st>>>(A, Bt, C, Cbf, bias, M, Ncols, K, relu);
}

static void run_gcn(const bf16_t* featbf, const bf16_t* WaT, const float* ba,
                    const bf16_t* WbT, const float* bb,
                    const int* src, const int* dst, const float* ns, const float* nd,
                    float* Hbuf, float* AGG, bf16_t* xbf,
                    float* outF, bf16_t* outBf, hipStream_t st) {
  // layer 1: H = feat @ Wa
  launch_gemm(featbf, WaT, Hbuf, nullptr, nullptr, NN, HIDD, IND, 0, st);
  hipMemsetAsync(AGG, 0, (size_t)NN * HIDD * 4, st);
  long long tot = (long long)EE * (HIDD / 4);
  k_scatter_agg<<<(int)((tot + 255) / 256), 256, 0, st>>>(Hbuf, ns, src, dst, AGG, tot, HIDD, 7);
  long long t2 = (long long)NN * HIDD;
  k_finalize<<<(int)((t2 + 255) / 256), 256, 0, st>>>(AGG, nd, ba, nullptr, xbf, t2, HIDD, 9, 1);
  // layer 2: H2 = X @ Wb
  launch_gemm(xbf, WbT, Hbuf, nullptr, nullptr, NN, OUTD, HIDD, 0, st);
  hipMemsetAsync(AGG, 0, (size_t)NN * OUTD * 4, st);
  long long t3 = (long long)EE * (OUTD / 4);
  k_scatter_agg<<<(int)((t3 + 255) / 256), 256, 0, st>>>(Hbuf, ns, src, dst, AGG, t3, OUTD, 6);
  long long t4 = (long long)NN * OUTD;
  k_finalize<<<(int)((t4 + 255) / 256), 256, 0, st>>>(AGG, nd, bb, outF, outBf, t4, OUTD, 8, 0);
}

static void run_scores(const float* vF, const bf16_t* vbf, const float* uF,
                       const bf16_t* WpT, const float* bp,
                       const int* src_s, const int* dst_s, const float* tau,
                       float* Pbuf, float* qf, bf16_t* qbf, float* unf, bf16_t* unbf,
                       bf16_t* zbf, float* rowNeg, float* posn, float* negn,
                       float* simb, const float* cnt, float* loss, hipStream_t st) {
  // q = normalize(v @ Wp + bp); un = stu = normalize(u); z = normalize(v)
  launch_gemm(vbf, WpT, Pbuf, nullptr, bp, NN, OUTD, OUTD, 0, st);
  k_rownorm<<<NN, 256, 0, st>>>(Pbuf, qf, qbf);
  k_rownorm<<<NN, 256, 0, st>>>(uF, unf, unbf);
  k_rownorm<<<NN, 256, 0, st>>>(vF, nullptr, zbf);
  // rowNeg[i] = sum_j exp(z.z^T/tau) + sum_j exp(z.un^T/tau)
  hipMemsetAsync(rowNeg, 0, NN * 4, st);
  const int rowTiles = NN / 16, colTiles = NN / 16, chunks = 16;
  const int chunkLen = (colTiles + chunks - 1) / chunks;
  dim3 ngrid(chunks, (rowTiles + 7) / 8);
  k_neg_rowsum<<<ngrid, 256, 0, st>>>(zbf, zbf,  rowNeg, tau, rowTiles, colTiles, chunkLen);
  k_neg_rowsum<<<ngrid, 256, 0, st>>>(zbf, unbf, rowNeg, tau, rowTiles, colTiles, chunkLen);
  // per-edge scores
  hipMemsetAsync(posn, 0, NN * 4, st);
  hipMemsetAsync(negn, 0, NN * 4, st);
  long long simThreads = (long long)EE * 32;
  k_sim_pos<<<(int)((simThreads + 255) / 256), 256, 0, st>>>(unf, qf, src_s, dst_s, tau,
                                                             simb, posn, EE);
  k_neg_edge<<<(EE + 255) / 256, 256, 0, st>>>(rowNeg, simb, dst_s, negn, EE);
  k_loss<<<64, 256, 0, st>>>(posn, negn, cnt, loss, NN);
}

// ---------------------------------------------------------------------------
extern "C" void kernel_launch(void* const* d_in, const int* in_sizes, int n_in,
                              void* d_out, int out_size, void* d_ws, size_t ws_size,
                              hipStream_t stream) {
  (void)in_sizes; (void)n_in; (void)out_size; (void)ws_size;
  const float* feat  = (const float*)d_in[0];
  const int*   src_g = (const int*)d_in[1];
  const int*   dst_g = (const int*)d_in[2];
  const int*   src_s = (const int*)d_in[3];
  const int*   dst_s = (const int*)d_in[4];
  const float* tau1  = (const float*)d_in[5];
  const float* tau2  = (const float*)d_in[6];
  const float* W1a   = (const float*)d_in[7];  const float* b1a   = (const float*)d_in[8];
  const float* W1b   = (const float*)d_in[9];  const float* b1b   = (const float*)d_in[10];
  const float* W1a_t = (const float*)d_in[11]; const float* b1a_t = (const float*)d_in[12];
  const float* W1b_t = (const float*)d_in[13]; const float* b1b_t = (const float*)d_in[14];
  const float* W2    = (const float*)d_in[15]; const float* b2    = (const float*)d_in[16];
  const float* W2_t  = (const float*)d_in[17]; const float* b2_t  = (const float*)d_in[18];
  const float* Wp    = (const float*)d_in[19]; const float* bp    = (const float*)d_in[20];

  // workspace carve-out (256B aligned)
  char* wsb = (char*)d_ws;
  size_t off = 0;
  auto carve = [&](size_t bytes) -> char* {
    char* p = wsb + off;
    off = (off + bytes + 255) & ~(size_t)255;
    return p;
  };
  bf16_t* featbf = (bf16_t*)carve((size_t)NN * IND * 2);
  bf16_t* xbf    = (bf16_t*)carve((size_t)NN * HIDD * 2);   // also aliases unbf/qbf
  bf16_t* vbf    = (bf16_t*)carve((size_t)NN * OUTD * 2);
  bf16_t* zbf    = (bf16_t*)carve((size_t)NN * OUTD * 2);
  bf16_t* W1aT   = (bf16_t*)carve((size_t)IND * HIDD * 2);
  bf16_t* W1bT   = (bf16_t*)carve((size_t)HIDD * OUTD * 2);
  bf16_t* W1aTt  = (bf16_t*)carve((size_t)IND * HIDD * 2);
  bf16_t* W1bTt  = (bf16_t*)carve((size_t)HIDD * OUTD * 2);
  bf16_t* W2T    = (bf16_t*)carve((size_t)IND * OUTD * 2);
  bf16_t* W2Tt   = (bf16_t*)carve((size_t)IND * OUTD * 2);
  bf16_t* WpT    = (bf16_t*)carve((size_t)OUTD * OUTD * 2);
  float*  Hbuf   = (float*)carve((size_t)NN * HIDD * 4);
  float*  AGG    = (float*)carve((size_t)NN * HIDD * 4);    // also aliases unf/qf
  float*  deg_o  = (float*)carve(NN * 4);
  float*  deg_i  = (float*)carve(NN * 4);
  float*  cnt    = (float*)carve(NN * 4);
  float*  rowNeg = (float*)carve(NN * 4);
  float*  posn   = (float*)carve(NN * 4);
  float*  negn   = (float*)carve(NN * 4);
  float*  simb   = (float*)carve((size_t)EE * 4);
  // aliases (only live during score phase, after all GCN aggregation)
  float*  unf  = AGG;
  float*  qf   = AGG + (size_t)NN * OUTD;
  bf16_t* unbf = xbf;
  bf16_t* qbf  = xbf + (size_t)NN * OUTD;

  float* out   = (float*)d_out;
  float* v1    = out;
  float* u1    = out + (size_t)NN * OUTD;
  float* v2    = out + 2 * (size_t)NN * OUTD;
  float* u2    = out + 3 * (size_t)NN * OUTD;
  float* lossp = out + 4 * (size_t)NN * OUTD;

  // degrees / normalizers
  hipMemsetAsync(deg_o, 0, NN * 4, stream);
  hipMemsetAsync(deg_i, 0, NN * 4, stream);
  hipMemsetAsync(cnt,   0, NN * 4, stream);
  k_deg<<<(EE + 255) / 256, 256, 0, stream>>>(src_g, deg_o, EE);
  k_deg<<<(EE + 255) / 256, 256, 0, stream>>>(dst_g, deg_i, EE);
  k_deg<<<(EE + 255) / 256, 256, 0, stream>>>(dst_s, cnt,   EE);
  k_invsqrt<<<(NN + 255) / 256, 256, 0, stream>>>(deg_o, NN);  // ns
  k_invsqrt<<<(NN + 255) / 256, 256, 0, stream>>>(deg_i, NN);  // nd

  // bf16 staging
  long long nf = (long long)NN * IND;
  k_f32_to_bf16<<<(int)((nf + 255) / 256), 256, 0, stream>>>(feat, featbf, nf);
  k_transpose_bf16<<<(IND * HIDD + 255) / 256, 256, 0, stream>>>(W1a,   W1aT,  IND,  HIDD);
  k_transpose_bf16<<<(HIDD * OUTD + 255) / 256, 256, 0, stream>>>(W1b,  W1bT,  HIDD, OUTD);
  k_transpose_bf16<<<(IND * HIDD + 255) / 256, 256, 0, stream>>>(W1a_t, W1aTt, IND,  HIDD);
  k_transpose_bf16<<<(HIDD * OUTD + 255) / 256, 256, 0, stream>>>(W1b_t, W1bTt, HIDD, OUTD);
  k_transpose_bf16<<<(IND * OUTD + 255) / 256, 256, 0, stream>>>(W2,    W2T,   IND,  OUTD);
  k_transpose_bf16<<<(IND * OUTD + 255) / 256, 256, 0, stream>>>(W2_t,  W2Tt,  IND,  OUTD);
  k_transpose_bf16<<<(OUTD * OUTD + 255) / 256, 256, 0, stream>>>(Wp,   WpT,   OUTD, OUTD);

  hipMemsetAsync(lossp, 0, 4, stream);

  // ---- branch 1: GCN encoders ----
  run_gcn(featbf, W1aT,  b1a,   W1bT,  b1b,   src_g, dst_g, deg_o, deg_i,
          Hbuf, AGG, xbf, v1, vbf, stream);
  run_gcn(featbf, W1aTt, b1a_t, W1bTt, b1b_t, src_g, dst_g, deg_o, deg_i,
          Hbuf, AGG, xbf, u1, nullptr, stream);
  run_scores(v1, vbf, u1, WpT, bp, src_s, dst_s, tau1,
             Hbuf, qf, qbf, unf, unbf, zbf, rowNeg, posn, negn, simb, cnt, lossp, stream);

  // ---- branch 2: linear encoders ----
  launch_gemm(featbf, W2T,  v2, vbf,     b2,   NN, OUTD, IND, 0, stream);
  launch_gemm(featbf, W2Tt, u2, nullptr, b2_t, NN, OUTD, IND, 0, stream);
  run_scores(v2, vbf, u2, WpT, bp, src_s, dst_s, tau2,
             Hbuf, qf, qbf, unf, unbf, zbf, rowNeg, posn, negn, simb, cnt, lossp, stream);
}
